// RelativeMultiHeadAttention_90280212562059
// MI455X (gfx1250) — matrix operand order repaired and checked by reading, not compiled, not run on hardware
//
#include <hip/hip_runtime.h>
#include <hip/hip_bf16.h>
#include <stdint.h>

typedef __bf16 bf16;
typedef __attribute__((ext_vector_type(16))) __bf16 v16bf;
typedef __attribute__((ext_vector_type(8)))  __bf16 v8bf;
typedef __attribute__((ext_vector_type(4)))  __bf16 v4bf;
typedef __attribute__((ext_vector_type(8)))  float  v8f;
typedef __attribute__((ext_vector_type(4)))  float  v4f;

#define QLEN   2048
#define BSZ    4
#define DMODEL 1024
#define NH     16
#define DH     64
#define DO     (NH * DH)      // 1024
#define MROWS  (QLEN * BSZ)   // 8192
#define N3     (3 * DO)       // 3072

// ---------------------------------------------------------------------------
// WMMA helpers (gfx1250, wave32): D(16x16,f32) = A(16x32,bf16) x B(32x16,bf16) + C
// ---------------------------------------------------------------------------
__device__ __forceinline__ v8f wmma_bf16(v16bf a, v16bf b, v8f c) {
  return __builtin_amdgcn_wmma_f32_16x16x32_bf16(
      /*neg_a=*/false, a, /*neg_b=*/false, b,
      /*c_mod=*/(short)0, c, /*reuse_a=*/false, /*reuse_b=*/false);
}

// A fragment from row-major [M,K] bf16 (ldk = row stride in elements).
// ISA layout: lane<16 -> row m=lane, K={0..7}u{16..23}; lane>=16 same rows,
// each half shifted +8. Two contiguous 16B loads per lane.
__device__ __forceinline__ v16bf load_a_frag(const bf16* __restrict__ base,
                                             int row0, int k0, int ldk, int lane) {
  const int m  = lane & 15;
  const int hi = lane >> 4;
  const bf16* p = base + (size_t)(row0 + m) * (size_t)ldk + (size_t)(k0 + hi * 8);
  union { v16bf v; v8bf h[2]; } u;
  u.h[0] = *(const v8bf*)(p);        // K = k0+8*hi    .. +7
  u.h[1] = *(const v8bf*)(p + 16);   // K = k0+16+8*hi .. +7
  return u.v;
}

// B fragment from B^T stored row-major [N,K] bf16.
// ISA layout: lane<16 -> col n=lane, K={0..7}u{16..23}; lane>=16 same cols, each half shifted +8.
__device__ __forceinline__ v16bf load_bt_frag(const bf16* __restrict__ base,
                                              int n0, int k0, int ldk, int lane) {
  const int n  = lane & 15;
  const int hi = lane >> 4;
  const bf16* p = base + (size_t)(n0 + n) * (size_t)ldk + (size_t)(k0 + hi * 8);
  union { v16bf v; v8bf h[2]; } u;
  u.h[0] = *(const v8bf*)(p);
  u.h[1] = *(const v8bf*)(p + 16);
  return u.v;
}

__device__ __forceinline__ v8f vzero8() {
  v8f z;
#pragma unroll
  for (int e = 0; e < 8; ++e) z[e] = 0.0f;
  return z;
}

// ---------------------------------------------------------------------------
// Pre-pass: f32 -> bf16 convert, and transpose-convert for weight matrices
// ---------------------------------------------------------------------------
__global__ __launch_bounds__(256) void cvt_bf16_kernel(const float* __restrict__ in,
                                                       bf16* __restrict__ out, int n) {
  int i = blockIdx.x * 256 + threadIdx.x;
  if (i < n) out[i] = (bf16)in[i];
}

// in: [rows][cols] f32  ->  out: [cols][rows] bf16
__global__ __launch_bounds__(256) void transpose_cvt_kernel(const float* __restrict__ in,
                                                            bf16* __restrict__ out,
                                                            int rows, int cols) {
  int i = blockIdx.x * 256 + threadIdx.x;
  if (i < rows * cols) {
    int r = i / cols, c = i - r * cols;
    out[(size_t)c * rows + r] = (bf16)in[i];
  }
}

// ---------------------------------------------------------------------------
// QKV projection: C[8192,3072] = Xbf[8192,1024] x Wqkv; epilogue splits q/k/v.
// Block = 256 thr = 8 waves; block tile 128x128; wave tile 64(M)x32(N).
// Section (q/k/v) and head index are uniform per 16-wide N-tile -> hoisted.
// ---------------------------------------------------------------------------
__global__ __launch_bounds__(256) void gemm_qkv_kernel(
    const bf16* __restrict__ X, const bf16* __restrict__ Wt,
    const float* __restrict__ bias_q, const float* __restrict__ bias_k,
    bf16* __restrict__ qA, bf16* __restrict__ qB,
    bf16* __restrict__ Km, bf16* __restrict__ Vt) {
  const int lane = threadIdx.x & 31;
  const int wid  = threadIdx.x >> 5;
  const int wm0  = blockIdx.y * 128 + (wid >> 2) * 64;
  const int wn0  = blockIdx.x * 128 + (wid & 3) * 32;

  v8f acc[4][2];
#pragma unroll
  for (int i = 0; i < 4; ++i)
#pragma unroll
    for (int j = 0; j < 2; ++j) acc[i][j] = vzero8();

  for (int kk = 0; kk < DMODEL; kk += 32) {
    v16bf a[4], b[2];
#pragma unroll
    for (int i = 0; i < 4; ++i) a[i] = load_a_frag(X, wm0 + 16 * i, kk, DMODEL, lane);
#pragma unroll
    for (int j = 0; j < 2; ++j) b[j] = load_bt_frag(Wt, wn0 + 16 * j, kk, DMODEL, lane);
    // prefetch next K chunk (speculative; dropped if OOB)
    __builtin_prefetch(X + (size_t)(wm0 + (lane & 15)) * DMODEL + kk + 32, 0, 3);
    __builtin_prefetch(Wt + (size_t)(wn0 + (lane & 15)) * DMODEL + kk + 32, 0, 3);
#pragma unroll
    for (int i = 0; i < 4; ++i)
#pragma unroll
      for (int j = 0; j < 2; ++j) acc[i][j] = wmma_bf16(a[i], b[j], acc[i][j]);
  }

  const int nl = lane & 15, hi = lane >> 4;
#pragma unroll
  for (int j = 0; j < 2; ++j) {
    const int ctile = wn0 + 16 * j;       // tile never crosses 64/1024 boundaries
    const int sec   = ctile >> 10;        // 0=q, 1=k, 2=v (wave-uniform)
    const int cc0   = ctile & 1023;
    const int h     = cc0 >> 6;
    const int d     = (cc0 & 63) + nl;
    const float bqv = (sec == 0) ? bias_q[cc0 + nl] : 0.0f;
    const float bkv = (sec == 0) ? bias_k[cc0 + nl] : 0.0f;
#pragma unroll
    for (int i = 0; i < 4; ++i) {
      const int r0 = wm0 + 16 * i + 8 * hi;
#pragma unroll
      for (int v = 0; v < 8; ++v) {
        const int r = r0 + v;              // flat row = q*BSZ + bb
        const int q = r >> 2, bb = r & 3;
        const float val = acc[i][j][v];
        const size_t idx = ((size_t)(bb * NH + h) * QLEN + q) * DH + d;
        if (sec == 0) {
          qA[idx] = (bf16)(val + bqv);
          qB[idx] = (bf16)(val + bkv);
        } else if (sec == 1) {
          Km[idx] = (bf16)val;
        } else {
          Vt[((size_t)(bb * NH + h) * DH + d) * QLEN + q] = (bf16)val;  // transposed
        }
      }
    }
  }
}

// ---------------------------------------------------------------------------
// r projection: C[2048,1024] = pos_bf x W_r + b_r -> Rm[h][p][d]
// ---------------------------------------------------------------------------
__global__ __launch_bounds__(256) void gemm_r_kernel(
    const bf16* __restrict__ P, const bf16* __restrict__ Wt,
    const float* __restrict__ br, bf16* __restrict__ Rm) {
  const int lane = threadIdx.x & 31;
  const int wid  = threadIdx.x >> 5;
  const int wm0  = blockIdx.y * 128 + (wid >> 2) * 64;
  const int wn0  = blockIdx.x * 128 + (wid & 3) * 32;

  v8f acc[4][2];
#pragma unroll
  for (int i = 0; i < 4; ++i)
#pragma unroll
    for (int j = 0; j < 2; ++j) acc[i][j] = vzero8();

  for (int kk = 0; kk < DMODEL; kk += 32) {
    v16bf a[4], b[2];
#pragma unroll
    for (int i = 0; i < 4; ++i) a[i] = load_a_frag(P, wm0 + 16 * i, kk, DMODEL, lane);
#pragma unroll
    for (int j = 0; j < 2; ++j) b[j] = load_bt_frag(Wt, wn0 + 16 * j, kk, DMODEL, lane);
#pragma unroll
    for (int i = 0; i < 4; ++i)
#pragma unroll
      for (int j = 0; j < 2; ++j) acc[i][j] = wmma_bf16(a[i], b[j], acc[i][j]);
  }

  const int nl = lane & 15, hi = lane >> 4;
#pragma unroll
  for (int j = 0; j < 2; ++j) {
    const int ctile = wn0 + 16 * j;
    const int h     = ctile >> 6;          // uniform per tile
    const int d     = (ctile & 63) + nl;
    const float bias = br[ctile + nl];
#pragma unroll
    for (int i = 0; i < 4; ++i) {
      const int p0 = wm0 + 16 * i + 8 * hi;
#pragma unroll
      for (int v = 0; v < 8; ++v) {
        Rm[((size_t)h * QLEN + (p0 + v)) * DH + d] = (bf16)(acc[i][j][v] + bias);
      }
    }
  }
}

// ---------------------------------------------------------------------------
// AC scores per (b,h): S[q,k] = (q+bias_q) . K  — GEMM 2048x2048x64
// ---------------------------------------------------------------------------
__global__ __launch_bounds__(256) void gemm_ac_kernel(
    const bf16* __restrict__ qA, const bf16* __restrict__ Km,
    float* __restrict__ S) {
  const int bh = blockIdx.z;
  const bf16* A  = qA + (size_t)bh * QLEN * DH;
  const bf16* Bt = Km + (size_t)bh * QLEN * DH;
  float* Sb = S + (size_t)bh * QLEN * QLEN;

  const int lane = threadIdx.x & 31;
  const int wid  = threadIdx.x >> 5;
  const int wm0  = blockIdx.y * 128 + (wid >> 2) * 64;
  const int wn0  = blockIdx.x * 128 + (wid & 3) * 32;

  v8f acc[4][2];
#pragma unroll
  for (int i = 0; i < 4; ++i)
#pragma unroll
    for (int j = 0; j < 2; ++j) acc[i][j] = vzero8();

#pragma unroll
  for (int kk = 0; kk < DH; kk += 32) {
    v16bf a[4], b[2];
#pragma unroll
    for (int i = 0; i < 4; ++i) a[i] = load_a_frag(A, wm0 + 16 * i, kk, DH, lane);
#pragma unroll
    for (int j = 0; j < 2; ++j) b[j] = load_bt_frag(Bt, wn0 + 16 * j, kk, DH, lane);
#pragma unroll
    for (int i = 0; i < 4; ++i)
#pragma unroll
      for (int j = 0; j < 2; ++j) acc[i][j] = wmma_bf16(a[i], b[j], acc[i][j]);
  }

  const int nl = lane & 15, hi = lane >> 4;
#pragma unroll
  for (int i = 0; i < 4; ++i)
#pragma unroll
    for (int j = 0; j < 2; ++j)
#pragma unroll
      for (int v = 0; v < 8; ++v) {
        const int q = wm0 + 16 * i + v + 8 * hi;
        const int k = wn0 + 16 * j + nl;
        Sb[(size_t)q * QLEN + k] = acc[i][j][v];
      }
}

// ---------------------------------------------------------------------------
// BD scores per (b,h): raw BD[q,p] = (q+bias_k) . r[p]; scatter with rel_shift.
// Bijective mapping (i==j): raw (q,p) -> (q, p+q-2047) if p+q>=2047
//                                     -> (q-1, p+q+1)  else (dropped if q==0)
// Unique writer per S element => plain read-add-write is race-free.
// ---------------------------------------------------------------------------
__global__ __launch_bounds__(256) void gemm_bd_kernel(
    const bf16* __restrict__ qB, const bf16* __restrict__ Rm,
    float* __restrict__ S) {
  const int bh = blockIdx.z;
  const int h  = bh & (NH - 1);
  const bf16* A  = qB + (size_t)bh * QLEN * DH;
  const bf16* Bt = Rm + (size_t)h * QLEN * DH;
  float* Sb = S + (size_t)bh * QLEN * QLEN;

  const int lane = threadIdx.x & 31;
  const int wid  = threadIdx.x >> 5;
  const int wm0  = blockIdx.y * 128 + (wid >> 2) * 64;
  const int wn0  = blockIdx.x * 128 + (wid & 3) * 32;

  v8f acc[4][2];
#pragma unroll
  for (int i = 0; i < 4; ++i)
#pragma unroll
    for (int j = 0; j < 2; ++j) acc[i][j] = vzero8();

#pragma unroll
  for (int kk = 0; kk < DH; kk += 32) {
    v16bf a[4], b[2];
#pragma unroll
    for (int i = 0; i < 4; ++i) a[i] = load_a_frag(A, wm0 + 16 * i, kk, DH, lane);
#pragma unroll
    for (int j = 0; j < 2; ++j) b[j] = load_bt_frag(Bt, wn0 + 16 * j, kk, DH, lane);
#pragma unroll
    for (int i = 0; i < 4; ++i)
#pragma unroll
      for (int j = 0; j < 2; ++j) acc[i][j] = wmma_bf16(a[i], b[j], acc[i][j]);
  }

  const int nl = lane & 15, hi = lane >> 4;
#pragma unroll
  for (int i = 0; i < 4; ++i)
#pragma unroll
    for (int j = 0; j < 2; ++j)
#pragma unroll
      for (int v = 0; v < 8; ++v) {
        const int q = wm0 + 16 * i + v + 8 * hi;
        const int p = wn0 + 16 * j + nl;
        const float val = acc[i][j][v];
        if (p + q >= QLEN - 1) {
          const int k = p + q - (QLEN - 1);
          float* t = &Sb[(size_t)q * QLEN + k];
          *t = *t + val;
        } else if (q >= 1) {
          const int k = p + q + 1;
          float* t = &Sb[(size_t)(q - 1) * QLEN + k];
          *t = *t + val;
        }
      }
}

// ---------------------------------------------------------------------------
// Row softmax of (AC+BD)/8; writes bf16 probabilities packed in-place at the
// start of each f32 score row (valid WMMA A-operand, ldk = 2*QLEN bf16 elems).
// One 256-thread block per row; vectorized b128 loads / b64 packed stores.
// ---------------------------------------------------------------------------
__global__ __launch_bounds__(256) void softmax_kernel(float* __restrict__ S) {
  const size_t row = blockIdx.x;
  float* r = S + row * QLEN;
  const int tid = threadIdx.x;
  const int lane = tid & 31, wv = tid >> 5;
  __shared__ float red[8];

  v4f x0 = *(const v4f*)(r + 4 * tid);
  v4f x1 = *(const v4f*)(r + (QLEN / 2) + 4 * tid);

  float m = -3.0e38f;
#pragma unroll
  for (int e = 0; e < 4; ++e) { m = fmaxf(m, x0[e]); m = fmaxf(m, x1[e]); }
#pragma unroll
  for (int off = 16; off > 0; off >>= 1) m = fmaxf(m, __shfl_xor(m, off, 32));
  if (lane == 0) red[wv] = m;
  __syncthreads();
  float bm = red[0];
#pragma unroll
  for (int w = 1; w < 8; ++w) bm = fmaxf(bm, red[w]);
  __syncthreads();

  float s = 0.0f;
#pragma unroll
  for (int e = 0; e < 4; ++e) {
    x0[e] = __expf((x0[e] - bm) * 0.125f); s += x0[e];
    x1[e] = __expf((x1[e] - bm) * 0.125f); s += x1[e];
  }
#pragma unroll
  for (int off = 16; off > 0; off >>= 1) s += __shfl_xor(s, off, 32);
  if (lane == 0) red[wv] = s;
  __syncthreads();
  float tot = 0.0f;
#pragma unroll
  for (int w = 0; w < 8; ++w) tot += red[w];
  const float inv = 1.0f / tot;

  // all f32 reads of this row happened before the first barrier -> safe overwrite
  bf16* pb = (bf16*)r;
  v4bf o0, o1;
#pragma unroll
  for (int e = 0; e < 4; ++e) {
    o0[e] = (bf16)(x0[e] * inv);
    o1[e] = (bf16)(x1[e] * inv);
  }
  *(v4bf*)(pb + 4 * tid) = o0;
  *(v4bf*)(pb + (QLEN / 2) + 4 * tid) = o1;
}

// ---------------------------------------------------------------------------
// ctx per (b,h): out = P(bf16, packed rows of S) x V  — GEMM 2048x64x2048.
// Block tile 256(M)x64(N): 8 waves as 4x2, wave tile 64x32.
// ---------------------------------------------------------------------------
__global__ __launch_bounds__(256) void gemm_ctx_kernel(
    const float* __restrict__ S, const bf16* __restrict__ Vt,
    float* __restrict__ out) {
  const int bh = blockIdx.z;
  const int bb = bh >> 4, h = bh & (NH - 1);
  const bf16* P  = (const bf16*)(S + (size_t)bh * QLEN * QLEN);  // ldk = 2*QLEN
  const bf16* Bt = Vt + (size_t)bh * DH * QLEN;                  // [64][2048]

  const int lane = threadIdx.x & 31;
  const int wid  = threadIdx.x >> 5;
  const int wm0  = blockIdx.y * 256 + (wid >> 1) * 64;
  const int wn0  = (wid & 1) * 32;

  v8f acc[4][2];
#pragma unroll
  for (int i = 0; i < 4; ++i)
#pragma unroll
    for (int j = 0; j < 2; ++j) acc[i][j] = vzero8();

  for (int kk = 0; kk < QLEN; kk += 32) {
    v16bf a[4], b[2];
#pragma unroll
    for (int i = 0; i < 4; ++i) a[i] = load_a_frag(P, wm0 + 16 * i, kk, 2 * QLEN, lane);
#pragma unroll
    for (int j = 0; j < 2; ++j) b[j] = load_bt_frag(Bt, wn0 + 16 * j, kk, QLEN, lane);
    __builtin_prefetch(P + (size_t)(wm0 + (lane & 15)) * (2 * QLEN) + kk + 32, 0, 3);
#pragma unroll
    for (int i = 0; i < 4; ++i)
#pragma unroll
      for (int j = 0; j < 2; ++j) acc[i][j] = wmma_bf16(a[i], b[j], acc[i][j]);
  }

  const int nl = lane & 15, hi = lane >> 4;
#pragma unroll
  for (int i = 0; i < 4; ++i)
#pragma unroll
    for (int j = 0; j < 2; ++j)
#pragma unroll
      for (int v = 0; v < 8; ++v) {
        const int q  = wm0 + 16 * i + v + 8 * hi;
        const int dl = wn0 + 16 * j + nl;                 // [0,64)
        out[((size_t)q * BSZ + bb) * DO + h * DH + dl] = acc[i][j][v];
      }
}

// ---------------------------------------------------------------------------
extern "C" void kernel_launch(void* const* d_in, const int* in_sizes, int n_in,
                              void* d_out, int out_size, void* d_ws, size_t ws_size,
                              hipStream_t stream) {
  (void)in_sizes; (void)n_in; (void)out_size; (void)ws_size;
  const float* word = (const float*)d_in[0];  // [2048,4,1024]
  const float* pos  = (const float*)d_in[1];  // [2048,1024]
  const float* bq   = (const float*)d_in[2];  // [16,64]
  const float* bk   = (const float*)d_in[3];  // [16,64]
  const float* Wqkv = (const float*)d_in[4];  // [1024,3072]
  const float* Wr   = (const float*)d_in[5];  // [1024,1024]
  const float* br   = (const float*)d_in[6];  // [1024]
  float* out = (float*)d_out;

  // Workspace carve-up (~96 MiB bf16 staging + 1 GiB f32 scores)
  char* base = (char*)d_ws;
  size_t o = 0;
  bf16* Xbf   = (bf16*)(base + o); o += (size_t)MROWS * DMODEL * 2;   // 16 MiB
  bf16* Pbf   = (bf16*)(base + o); o += (size_t)QLEN * DMODEL * 2;    //  4 MiB
  bf16* Wqkvt = (bf16*)(base + o); o += (size_t)N3 * DMODEL * 2;      //  6 MiB
  bf16* Wrt   = (bf16*)(base + o); o += (size_t)DO * DMODEL * 2;      //  2 MiB
  bf16* qA    = (bf16*)(base + o); o += (size_t)BSZ * NH * QLEN * DH * 2;  // 16 MiB
  bf16* qB    = (bf16*)(base + o); o += (size_t)BSZ * NH * QLEN * DH * 2;
  bf16* Km    = (bf16*)(base + o); o += (size_t)BSZ * NH * QLEN * DH * 2;
  bf16* Vt    = (bf16*)(base + o); o += (size_t)BSZ * NH * QLEN * DH * 2;
  bf16* Rm    = (bf16*)(base + o); o += (size_t)NH * QLEN * DH * 2;   //  4 MiB
  float* S    = (float*)(base + o);                                   //  1 GiB

  const dim3 blk(256);

  // 1) convert / transpose staging
  cvt_bf16_kernel<<<dim3((MROWS * DMODEL + 255) / 256), blk, 0, stream>>>(word, Xbf, MROWS * DMODEL);
  cvt_bf16_kernel<<<dim3((QLEN * DMODEL + 255) / 256), blk, 0, stream>>>(pos, Pbf, QLEN * DMODEL);
  transpose_cvt_kernel<<<dim3((DMODEL * N3 + 255) / 256), blk, 0, stream>>>(Wqkv, Wqkvt, DMODEL, N3);
  transpose_cvt_kernel<<<dim3((DMODEL * DO + 255) / 256), blk, 0, stream>>>(Wr, Wrt, DMODEL, DO);

  // 2) projections
  gemm_qkv_kernel<<<dim3(N3 / 128, MROWS / 128), blk, 0, stream>>>(Xbf, Wqkvt, bq, bk, qA, qB, Km, Vt);
  gemm_r_kernel<<<dim3(DO / 128, QLEN / 128), blk, 0, stream>>>(Pbf, Wrt, br, Rm);

  // 3) scores: AC (plain store) then BD (rel-shift scatter add)
  gemm_ac_kernel<<<dim3(QLEN / 128, QLEN / 128, BSZ * NH), blk, 0, stream>>>(qA, Km, S);
  gemm_bd_kernel<<<dim3(QLEN / 128, QLEN / 128, BSZ * NH), blk, 0, stream>>>(qB, Rm, S);

  // 4) softmax (scale 1/8 folded in), probabilities packed bf16 in-place
  softmax_kernel<<<dim3((unsigned)(BSZ * NH * QLEN)), blk, 0, stream>>>(S);

  // 5) ctx = P x V, scattered to [qlen, b, H*DH]
  gemm_ctx_kernel<<<dim3(1, QLEN / 256, BSZ * NH), blk, 0, stream>>>(S, Vt, out);
}
